// FusedAttn_57775900066023
// MI455X (gfx1250) — compile-verified, matrix operand order checked by reading, not run on hardware
//
#include <hip/hip_runtime.h>
#include <cstdint>
#include <cstddef>

typedef unsigned short bf16_t;
typedef __attribute__((ext_vector_type(16))) __bf16 v16bf;
typedef __attribute__((ext_vector_type(8)))  float  v8f;

#define H_   4096
#define EH_  1024
#define NH_  32
#define ENH_ 16
#define HD_  128
#define EHD_ 64
#define AH_  1024
#define B_   2
#define S_   1024
#define T_   (B_*S_)     /* 2048 tokens  */
#define MH_  (T_*NH_)    /* 65536 token-heads */
#define CH_  16384       /* adapter row chunk */

// ---------------------------------------------------------------- utilities
__device__ __forceinline__ bf16_t f2bf(float f) {
  union { float f; unsigned u; } c; c.f = f;
  unsigned u = c.u;
  u += 0x7FFFu + ((u >> 16) & 1u);   // round-to-nearest-even
  return (bf16_t)(u >> 16);
}

__device__ __forceinline__ v16bf load_frag16(const bf16_t* p) {
  union { v16bf v; uint4 q[2]; } u;
  u.q[0] = *(const uint4*)(p);
  u.q[1] = *(const uint4*)(p + 8);
  return u.v;
}

__device__ __forceinline__ float gelu_tanh(float x) {
  float x3 = x * x * x;
  return 0.5f * x * (1.0f + tanhf(0.7978845608028654f * (x + 0.044715f * x3)));
}

// ---------------------------------------------------------------- TDM support
#if __has_builtin(__builtin_amdgcn_tensor_load_to_lds) && __has_builtin(__builtin_amdgcn_s_wait_tensorcnt)
#define USE_TDM 1
typedef __attribute__((ext_vector_type(4))) unsigned int u32x4;
typedef __attribute__((ext_vector_type(8))) int i32x8;
typedef __attribute__((ext_vector_type(4))) int i32x4;

// 2D tile DMA: copy tileD1 rows x tileD0 elements (bf16) from row-major
// global (row stride = strideElems) into contiguous LDS at lds_off.
__device__ __forceinline__ void tdm_load_2d(const bf16_t* gsrc, unsigned lds_off,
                                            int tileD0, int tileD1, int strideElems) {
  unsigned long long ga = (unsigned long long)(size_t)gsrc;
  u32x4 g0;
  g0[0] = 1u;                                        // count=1, user descriptor
  g0[1] = lds_off;                                   // LDS byte address
  g0[2] = (unsigned)(ga & 0xFFFFFFFFu);              // global_addr[31:0]
  g0[3] = (unsigned)((ga >> 32) & 0x01FFFFFFu) | 0x80000000u;  // addr[56:32] | type=2
  unsigned td0 = (unsigned)strideElems;              // tensor_dim0
  unsigned td1 = (unsigned)tileD1;                   // tensor_dim1
  i32x8 g1;
  g1[0] = 0x00010000;                                // data_size=1 (2 bytes)
  g1[1] = (int)((td0 & 0xFFFFu) << 16);              // tensor_dim0[15:0] @ bits 63:48
  g1[2] = (int)(((td0 >> 16) & 0xFFFFu) | ((td1 & 0xFFFFu) << 16));
  g1[3] = (int)((((unsigned)tileD0 & 0xFFFFu) << 16) | ((td1 >> 16) & 0xFFFFu));
  g1[4] = (int)((unsigned)tileD1 & 0xFFFFu);         // tile_dim1, tile_dim2=0
  g1[5] = (int)(unsigned)strideElems;                // tensor_dim0_stride[31:0]
  g1[6] = 0;
  g1[7] = 0;
  i32x4 z4 = {0, 0, 0, 0};
#if __clang_major__ >= 23
  i32x8 z8 = {0, 0, 0, 0, 0, 0, 0, 0};
  __builtin_amdgcn_tensor_load_to_lds(g0, g1, z4, z4, z8, 0);
#else
  __builtin_amdgcn_tensor_load_to_lds(g0, g1, z4, z4, 0);
#endif
}
#else
#define USE_TDM 0
#endif

// ---------------------------------------------------------------- async global->LDS
#if __has_builtin(__builtin_amdgcn_global_load_async_to_lds_b128) && __has_builtin(__builtin_amdgcn_s_wait_asynccnt)
#define USE_ASYNC 1
typedef int v4i32 __attribute__((vector_size(16)));
__device__ __forceinline__ void async_copy16(const void* g, void* l) {
  __builtin_amdgcn_global_load_async_to_lds_b128((v4i32*)g, (v4i32*)l, 0, 0);
}
#else
#define USE_ASYNC 0
#endif

// ---------------------------------------------------------------- cvt fp32->bf16
__global__ void cvt_bf16_kernel(const float* __restrict__ in, bf16_t* __restrict__ out,
                                size_t n) {
  size_t i = (size_t)blockIdx.x * 256 + threadIdx.x;
  if (i < n) out[i] = f2bf(in[i]);
}

// W[K,N] fp32 -> Wt[N,K] bf16
__global__ void transpose_cvt_kernel(const float* __restrict__ W, bf16_t* __restrict__ Wt,
                                     int K, int N) {
  size_t i = (size_t)blockIdx.x * 256 + threadIdx.x;
  size_t tot = (size_t)K * N;
  if (i < tot) {
    int n = (int)(i / K), k = (int)(i % K);
    Wt[i] = f2bf(W[(size_t)k * N + n]);
  }
}

// ---------------------------------------------------------------- RMSNorm -> bf16
__global__ __launch_bounds__(256)
void rmsnorm_kernel(const float* __restrict__ x, const float* __restrict__ w,
                    bf16_t* __restrict__ out, int L) {
  __shared__ float red[8];
  int row = blockIdx.x;
  const float* xr = x + (size_t)row * L;
  float s = 0.f;
  for (int i = threadIdx.x; i < L; i += 256) { float v = xr[i]; s += v * v; }
  for (int o = 16; o > 0; o >>= 1) s += __shfl_xor(s, o, 32);
  if ((threadIdx.x & 31) == 0) red[threadIdx.x >> 5] = s;
  __syncthreads();
  float tot = 0.f;
  for (int i = 0; i < 8; i++) tot += red[i];
  float inv = rsqrtf(tot / (float)L + 1e-6f);
  for (int i = threadIdx.x; i < L; i += 256)
    out[(size_t)row * L + i] = f2bf(xr[i] * inv * w[i]);
}

// ---------------------------------------------------------------- WMMA GEMM
// C[M,N] = A[M,K](bf16) x Bt[N,K](bf16)^T  (+bias, +gelu, +residual)
// Double-buffered LDS; A via async global->LDS (or reg pipeline), B via TDM.
#define BM 128
#define BK 32
template<int TBN>
__global__ __launch_bounds__(256)
void gemm_bf16_kernel(const bf16_t* __restrict__ A, const bf16_t* __restrict__ Bt,
                      const float* __restrict__ bias, const float* __restrict__ residual,
                      float* __restrict__ Cf, bf16_t* __restrict__ Cb,
                      int M, int N, int K, int act) {
  constexpr int NF = TBN / 32;                   // N frags per wave (2 or 4)
  constexpr int NB = (TBN * BK) / (8 * 256);     // B uint4 chunks per thread (1 or 2)
  __shared__ bf16_t As[2][BM * BK];              // 2 x 8 KB
  __shared__ bf16_t Bs[2][TBN * BK];             // 2 x (4|8) KB
  const int tid  = threadIdx.x;
  const int lane = tid & 31;
  const int wave = tid >> 5;                     // 8 waves: 4 along M x 2 along N
  const int wm   = wave & 3;
  const int wn   = wave >> 2;
  const int row0 = blockIdx.y * BM;
  const int col0 = blockIdx.x * TBN;
  const int la   = lane & 15;
  const int hs   = (lane >> 4) * 16;
  const int ar0  = tid >> 2;                     // A chunk rows (2 per thread)
  const int ar1  = (tid + 256) >> 2;
  const int ak   = (tid & 3) * 8;

  v8f acc[2][NF];
#pragma unroll
  for (int mf = 0; mf < 2; mf++)
#pragma unroll
    for (int nf = 0; nf < NF; nf++)
#pragma unroll
      for (int v = 0; v < 8; v++) acc[mf][nf][v] = 0.f;

#if !USE_ASYNC
  uint4 aReg[2];
  uint4 bReg[NB];
#endif

  // ---- preload tile 0
#if USE_ASYNC
  async_copy16(&A[(size_t)(row0 + ar0) * K + ak], &As[0][ar0 * BK + ak]);
  async_copy16(&A[(size_t)(row0 + ar1) * K + ak], &As[0][ar1 * BK + ak]);
#pragma unroll
  for (int i = 0; i < NB; i++) {
    int c = tid + i * 256;
    async_copy16(&Bt[(size_t)(col0 + (c >> 2)) * K + (c & 3) * 8],
                 &Bs[0][(c >> 2) * BK + (c & 3) * 8]);
  }
  __builtin_amdgcn_s_wait_asynccnt(0);
#else
  aReg[0] = *(const uint4*)&A[(size_t)(row0 + ar0) * K + ak];
  aReg[1] = *(const uint4*)&A[(size_t)(row0 + ar1) * K + ak];
#pragma unroll
  for (int i = 0; i < NB; i++) {
    int c = tid + i * 256;
    bReg[i] = *(const uint4*)&Bt[(size_t)(col0 + (c >> 2)) * K + (c & 3) * 8];
  }
  *(uint4*)&As[0][ar0 * BK + ak] = aReg[0];
  *(uint4*)&As[0][ar1 * BK + ak] = aReg[1];
#pragma unroll
  for (int i = 0; i < NB; i++) {
    int c = tid + i * 256;
    *(uint4*)&Bs[0][(c >> 2) * BK + (c & 3) * 8] = bReg[i];
  }
#endif
  __syncthreads();

  int buf = 0;
  for (int kk = 0; kk < K; kk += BK) {
    const bool nxt = (kk + BK) < K;
    if (nxt) {
#if USE_ASYNC
      async_copy16(&A[(size_t)(row0 + ar0) * K + kk + BK + ak],
                   &As[buf ^ 1][ar0 * BK + ak]);
      async_copy16(&A[(size_t)(row0 + ar1) * K + kk + BK + ak],
                   &As[buf ^ 1][ar1 * BK + ak]);
#else
      aReg[0] = *(const uint4*)&A[(size_t)(row0 + ar0) * K + kk + BK + ak];
      aReg[1] = *(const uint4*)&A[(size_t)(row0 + ar1) * K + kk + BK + ak];
#endif
#if USE_TDM
      if (tid < 32)   // one DMA per block: wave 0 issues TDM into the back buffer
        tdm_load_2d(&Bt[(size_t)col0 * K + kk + BK],
                    (unsigned)(size_t)&Bs[buf ^ 1][0], BK, TBN, K);
#elif USE_ASYNC
#pragma unroll
      for (int i = 0; i < NB; i++) {
        int c = tid + i * 256;
        async_copy16(&Bt[(size_t)(col0 + (c >> 2)) * K + kk + BK + (c & 3) * 8],
                     &Bs[buf ^ 1][(c >> 2) * BK + (c & 3) * 8]);
      }
#else
#pragma unroll
      for (int i = 0; i < NB; i++) {
        int c = tid + i * 256;
        bReg[i] = *(const uint4*)&Bt[(size_t)(col0 + (c >> 2)) * K + kk + BK + (c & 3) * 8];
      }
#endif
    }

    // ---- compute current tile: 2 x NF WMMAs
    v16bf af[2], bfv[NF];
#pragma unroll
    for (int mf = 0; mf < 2; mf++)
      af[mf] = load_frag16(&As[buf][(wm * 32 + mf * 16 + la) * BK + hs]);
#pragma unroll
    for (int nf = 0; nf < NF; nf++)
      bfv[nf] = load_frag16(&Bs[buf][(wn * (TBN / 2) + nf * 16 + la) * BK + hs]);
#pragma unroll
    for (int mf = 0; mf < 2; mf++)
#pragma unroll
      for (int nf = 0; nf < NF; nf++)
        acc[mf][nf] = __builtin_amdgcn_wmma_f32_16x16x32_bf16(
            false, af[mf], false, bfv[nf], (short)0, acc[mf][nf], false, false);

    if (nxt) {
#if USE_ASYNC
      __builtin_amdgcn_s_wait_asynccnt(0);
#else
      *(uint4*)&As[buf ^ 1][ar0 * BK + ak] = aReg[0];
      *(uint4*)&As[buf ^ 1][ar1 * BK + ak] = aReg[1];
#if !USE_TDM
#pragma unroll
      for (int i = 0; i < NB; i++) {
        int c = tid + i * 256;
        *(uint4*)&Bs[buf ^ 1][(c >> 2) * BK + (c & 3) * 8] = bReg[i];
      }
#endif
#endif
#if USE_TDM
      if (tid < 32) __builtin_amdgcn_s_wait_tensorcnt(0);
#endif
    }
    __syncthreads();
    buf ^= 1;
  }

  // ---- epilogue: C layout — VGPR v: lanes 0-15 row v, lanes 16-31 row v+8
  const int mh = (lane >> 4) * 8;
#pragma unroll
  for (int mf = 0; mf < 2; mf++)
#pragma unroll
    for (int nf = 0; nf < NF; nf++)
#pragma unroll
      for (int v = 0; v < 8; v++) {
        int r = row0 + wm * 32 + mf * 16 + mh + v;
        int c = col0 + wn * (TBN / 2) + nf * 16 + la;
        float x = acc[mf][nf][v];
        if (bias)     x += bias[c];
        if (act)      x  = gelu_tanh(x);
        if (residual) x += residual[(size_t)r * N + c];
        if (Cf) Cf[(size_t)r * N + c] = x;
        if (Cb) Cb[(size_t)r * N + c] = f2bf(x);
      }
}

// ---------------------------------------------------------------- entity head replicate
__global__ void replicate_kernel(const float* __restrict__ e, bf16_t* __restrict__ erep) {
  size_t i = (size_t)blockIdx.x * 256 + threadIdx.x;   // over T*NH*EHD
  if (i >= (size_t)MH_ * EHD_) return;
  int d   = (int)(i & 63);
  int h   = (int)((i >> 6) & 31);
  int tok = (int)(i >> 11);
  erep[i] = f2bf(e[(size_t)tok * EH_ + (h & 15) * EHD_ + d]);
}

// ---------------------------------------------------------------- merge + RoPE + pack
__global__ __launch_bounds__(128)
void merge_rope_kernel(const float* __restrict__ t, const float* __restrict__ a2,
                       const int* __restrict__ ttm, const int* __restrict__ pos,
                       bf16_t* __restrict__ outp, int do_rope) {
  int tok = blockIdx.x, h = blockIdx.y, d = threadIdx.x;
  int b = tok >> 10, s = tok & (S_ - 1);
  bool ent = (ttm[tok] == 1);
  float v = ent ? a2[((size_t)tok * NH_ + h) * HD_ + d]
                : t[(size_t)tok * H_ + h * HD_ + d];
  if (do_rope) {
    int dp = (d < 64) ? d + 64 : d - 64;
    float vp = ent ? a2[((size_t)tok * NH_ + h) * HD_ + dp]
                   : t[(size_t)tok * H_ + h * HD_ + dp];
    float rot = (d < 64) ? -vp : vp;
    float p   = (float)pos[tok];
    int   i   = d & 63;
    float inv = __expf(-(float)i * (9.210340371976184f / 64.f));  // 10000^{-i/64}
    float cs  = __cosf(p * inv), sn = __sinf(p * inv);
    v = v * cs + rot * sn;
  }
  outp[(((size_t)b * NH_ + h) * S_ + s) * HD_ + d] = f2bf(v);
}

// ---------------------------------------------------------------- flash attention (WMMA)
__global__ __launch_bounds__(128)
void flash_attn_kernel(const bf16_t* __restrict__ Q, const bf16_t* __restrict__ Kg,
                       const bf16_t* __restrict__ Vg, bf16_t* __restrict__ O) {
  __shared__ bf16_t Ks[32 * HD_];        // [key][d]    8 KB
  __shared__ bf16_t Vs[HD_ * 32];        // [d][key]    8 KB (transposed)
  __shared__ bf16_t Ps[4][16 * 32];      // per-wave P  4 KB
  const int tid  = threadIdx.x;
  const int lane = tid & 31;
  const int wave = tid >> 5;             // 4 waves; each owns 16 q rows
  const int qt   = blockIdx.x;
  const int h    = blockIdx.y;
  const int b    = blockIdx.z;
  const size_t base = ((size_t)b * NH_ + h) * S_ * HD_;
  const int q0 = qt * 64 + wave * 16;
  const int la = lane & 15;
  const int hi = lane >> 4;
  const float scale = 0.08838834764831845f;   // 1/sqrt(128)

  v16bf qf[4];
#pragma unroll
  for (int c = 0; c < 4; c++)
    qf[c] = load_frag16(Q + base + (size_t)(q0 + la) * HD_ + c * 32 + hi * 16);

  v8f oacc[8];
#pragma unroll
  for (int nf = 0; nf < 8; nf++)
#pragma unroll
    for (int v = 0; v < 8; v++) oacc[nf][v] = 0.f;
  float m[8], l[8];
#pragma unroll
  for (int v = 0; v < 8; v++) { m[v] = -1e30f; l[v] = 0.f; }

  const int kend = qt * 64 + 64;
  for (int j = 0; j < kend; j += 32) {
#pragma unroll
    for (int i = 0; i < 4; i++) {
      int c   = tid + i * 128;
      int key = c >> 4;
      int d8  = (c & 15) * 8;
#if USE_ASYNC
      async_copy16(&Kg[base + (size_t)(j + key) * HD_ + d8], &Ks[key * HD_ + d8]);
#else
      *(uint4*)&Ks[key * HD_ + d8] =
          *(const uint4*)&Kg[base + (size_t)(j + key) * HD_ + d8];
#endif
      union { uint4 q; bf16_t e[8]; } u;
      u.q = *(const uint4*)&Vg[base + (size_t)(j + key) * HD_ + d8];
#pragma unroll
      for (int e = 0; e < 8; e++) Vs[(d8 + e) * 32 + key] = u.e[e];
    }
#if USE_ASYNC
    __builtin_amdgcn_s_wait_asynccnt(0);
#endif
    __syncthreads();

    v8f sacc[2];
#pragma unroll
    for (int nf = 0; nf < 2; nf++)
#pragma unroll
      for (int v = 0; v < 8; v++) sacc[nf][v] = 0.f;
#pragma unroll
    for (int c = 0; c < 4; c++) {
#pragma unroll
      for (int nf = 0; nf < 2; nf++) {
        v16bf kf = load_frag16(&Ks[(nf * 16 + la) * HD_ + c * 32 + hi * 16]);
        sacc[nf] = __builtin_amdgcn_wmma_f32_16x16x32_bf16(
            false, qf[c], false, kf, (short)0, sacc[nf], false, false);
      }
    }

#pragma unroll
    for (int v = 0; v < 8; v++) {
      int row = q0 + v + hi * 8;
      float s0 = sacc[0][v] * scale;
      float s1 = sacc[1][v] * scale;
      if (j + la > row)      s0 = -1e30f;
      if (j + 16 + la > row) s1 = -1e30f;
      float rmax = fmaxf(s0, s1);
#pragma unroll
      for (int o = 1; o <= 8; o <<= 1) rmax = fmaxf(rmax, __shfl_xor(rmax, o, 32));
      float mnew = fmaxf(m[v], rmax);
      float p0 = __expf(s0 - mnew), p1 = __expf(s1 - mnew);
      float ps = p0 + p1;
#pragma unroll
      for (int o = 1; o <= 8; o <<= 1) ps += __shfl_xor(ps, o, 32);
      float alpha = __expf(m[v] - mnew);
      l[v] = l[v] * alpha + ps;
      m[v] = mnew;
#pragma unroll
      for (int nf = 0; nf < 8; nf++) oacc[nf][v] *= alpha;
      Ps[wave][(v + hi * 8) * 32 + la]      = f2bf(p0);
      Ps[wave][(v + hi * 8) * 32 + 16 + la] = f2bf(p1);
    }
    __syncthreads();

    v16bf pf = load_frag16(&Ps[wave][la * 32 + hi * 16]);
#pragma unroll
    for (int nf = 0; nf < 8; nf++) {
      v16bf vf = load_frag16(&Vs[(nf * 16 + la) * 32 + hi * 16]);
      oacc[nf] = __builtin_amdgcn_wmma_f32_16x16x32_bf16(
          false, pf, false, vf, (short)0, oacc[nf], false, false);
    }
    __syncthreads();
  }

#pragma unroll
  for (int nf = 0; nf < 8; nf++)
#pragma unroll
    for (int v = 0; v < 8; v++) {
      int q = q0 + v + hi * 8;
      int d = nf * 16 + la;
      float val = oacc[nf][v] / l[v];
      O[(size_t)(b * S_ + q) * H_ + h * HD_ + d] = f2bf(val);
    }
}

// ---------------------------------------------------------------- replica mean
__global__ void mean_rep_kernel(const float* __restrict__ e1, bf16_t* __restrict__ em) {
  size_t i = (size_t)blockIdx.x * 256 + threadIdx.x;   // over T*EH
  if (i >= (size_t)T_ * EH_) return;
  int d   = (int)(i & 63);
  int enh = (int)((i >> 6) & 15);
  int tok = (int)(i >> 10);
  float a = e1[(size_t)tok * (NH_ * EHD_) + enh * EHD_ + d];
  float b = e1[(size_t)tok * (NH_ * EHD_) + (enh + 16) * EHD_ + d];
  em[i] = f2bf(0.5f * (a + b));
}

// ---------------------------------------------------------------- final merge
__global__ void final_merge_kernel(const float* __restrict__ outt,
                                   const float* __restrict__ oute,
                                   const int* __restrict__ ttm,
                                   float* __restrict__ out) {
  size_t i = (size_t)blockIdx.x * 256 + threadIdx.x;
  if (i >= (size_t)T_ * H_) return;
  int tok = (int)(i >> 12);
  out[i] = (ttm[tok] == 1) ? oute[i] : outt[i];
}

// ================================================================ host
static inline void launch_gemm(const bf16_t* A, const bf16_t* Bt, const float* bias,
                               const float* res, float* Cf, bf16_t* Cb,
                               int M, int N, int K, int act, hipStream_t s) {
  if (N % 128 == 0) {
    dim3 g(N / 128, M / BM);
    gemm_bf16_kernel<128><<<g, dim3(256), 0, s>>>(A, Bt, bias, res, Cf, Cb, M, N, K, act);
  } else {
    dim3 g(N / 64, M / BM);
    gemm_bf16_kernel<64><<<g, dim3(256), 0, s>>>(A, Bt, bias, res, Cf, Cb, M, N, K, act);
  }
}

extern "C" void kernel_launch(void* const* d_in, const int* in_sizes, int n_in,
                              void* d_out, int out_size, void* d_ws, size_t ws_size,
                              hipStream_t stream) {
  (void)in_sizes; (void)n_in; (void)out_size; (void)ws_size;

  const float* x   = (const float*)d_in[0];
  const int*   ttm = (const int*)d_in[1];
  const int*   pos = (const int*)d_in[2];
  // d_in[3] attention_mask: exactly causal -> applied analytically in flash kernel

  size_t off = 0;
  auto alloc = [&](size_t bytes) -> void* {
    off = (off + 255) & ~(size_t)255;
    void* p = (char*)d_ws + off;
    off += bytes;
    return p;
  };

  // bf16 transposed weights
  bf16_t* w_ad1t = (bf16_t*)alloc((size_t)EH_ * H_ * 2);
  bf16_t* w_ad2t = (bf16_t*)alloc((size_t)EH_ * EH_ * 2);
  bf16_t *twt[3], *ewt[3], *a1t[3], *a2t[3];
  for (int n = 0; n < 3; n++) {
    twt[n] = (bf16_t*)alloc((size_t)H_ * H_ * 2);
    ewt[n] = (bf16_t*)alloc((size_t)EH_ * EH_ * 2);
    a1t[n] = (bf16_t*)alloc((size_t)AH_ * EHD_ * 2);
    a2t[n] = (bf16_t*)alloc((size_t)HD_ * AH_ * 2);
  }
  bf16_t* otwt  = (bf16_t*)alloc((size_t)H_ * H_ * 2);
  bf16_t* oeint = (bf16_t*)alloc((size_t)EHD_ * HD_ * 2);
  bf16_t* oewt  = (bf16_t*)alloc((size_t)EH_ * EH_ * 2);
  bf16_t* oeot  = (bf16_t*)alloc((size_t)H_ * EH_ * 2);

  // activations
  bf16_t* x_bf    = (bf16_t*)alloc((size_t)T_ * H_ * 2);
  bf16_t* h1_bf   = (bf16_t*)alloc((size_t)T_ * EH_ * 2);
  float*  xe      = (float*) alloc((size_t)T_ * EH_ * 4);
  bf16_t* xe_bf   = (bf16_t*)alloc((size_t)T_ * EH_ * 2);
  bf16_t* tn_bf   = (bf16_t*)alloc((size_t)T_ * H_ * 2);
  bf16_t* en_bf   = (bf16_t*)alloc((size_t)T_ * EH_ * 2);
  float*  tbuf    = (float*) alloc((size_t)T_ * H_ * 4);
  float*  ebuf    = (float*) alloc((size_t)T_ * EH_ * 4);
  bf16_t* erep_bf = (bf16_t*)alloc((size_t)MH_ * EHD_ * 2);
  bf16_t* a1_bf   = (bf16_t*)alloc((size_t)CH_ * AH_ * 2);
  float*  a2buf   = (float*) alloc((size_t)MH_ * HD_ * 4);
  bf16_t* qkv_bf[3];
  for (int n = 0; n < 3; n++) qkv_bf[n] = (bf16_t*)alloc((size_t)T_ * H_ * 2);
  bf16_t* o_bf    = (bf16_t*)alloc((size_t)T_ * H_ * 2);
  float*  outt    = (float*) alloc((size_t)T_ * H_ * 4);
  float*  e1      = (float*) alloc((size_t)MH_ * EHD_ * 4);
  bf16_t* em_bf   = (bf16_t*)alloc((size_t)T_ * EH_ * 2);
  bf16_t* e2_bf   = (bf16_t*)alloc((size_t)T_ * EH_ * 2);
  float*  oute    = (float*) alloc((size_t)T_ * H_ * 4);

  auto tcvt = [&](int idx, bf16_t* dst, int K, int N) {
    size_t tot = (size_t)K * N;
    transpose_cvt_kernel<<<dim3((unsigned)((tot + 255) / 256)), dim3(256), 0, stream>>>(
        (const float*)d_in[idx], dst, K, N);
  };

  // ---- weight prep (transpose + bf16)
  tcvt(4, w_ad1t, H_, EH_);
  tcvt(6, w_ad2t, EH_, EH_);
  for (int n = 0; n < 3; n++) {
    int pb = 8 + n * 10;
    tcvt(pb + 1, twt[n], H_, H_);
    tcvt(pb + 4, ewt[n], EH_, EH_);
    tcvt(pb + 6, a1t[n], EHD_, AH_);
    tcvt(pb + 8, a2t[n], AH_, HD_);
  }
  tcvt(38, otwt, H_, H_);
  tcvt(40, oeint, HD_, EHD_);
  tcvt(42, oewt, EH_, EH_);
  tcvt(44, oeot, EH_, H_);

  // ---- x -> bf16
  {
    size_t n = (size_t)T_ * H_;
    cvt_bf16_kernel<<<dim3((unsigned)((n + 255) / 256)), dim3(256), 0, stream>>>(x, x_bf, n);
  }

  // ---- input adapter: x_e = W2(gelu(W1 x + b1)) + b2
  launch_gemm(x_bf, w_ad1t, (const float*)d_in[5], nullptr, nullptr, h1_bf,
              T_, EH_, H_, 1, stream);
  launch_gemm(h1_bf, w_ad2t, (const float*)d_in[7], nullptr, xe, xe_bf,
              T_, EH_, EH_, 0, stream);

  // ---- q/k/v projections
  for (int n = 0; n < 3; n++) {
    int pb = 8 + n * 10;
    rmsnorm_kernel<<<dim3(T_), dim3(256), 0, stream>>>(x, (const float*)d_in[pb + 0], tn_bf, H_);
    rmsnorm_kernel<<<dim3(T_), dim3(256), 0, stream>>>(xe, (const float*)d_in[pb + 3], en_bf, EH_);
    launch_gemm(tn_bf, twt[n], (const float*)d_in[pb + 2], nullptr, tbuf, nullptr,
                T_, H_, H_, 0, stream);
    launch_gemm(en_bf, ewt[n], (const float*)d_in[pb + 5], nullptr, ebuf, nullptr,
                T_, EH_, EH_, 0, stream);
    {
      size_t tot = (size_t)MH_ * EHD_;
      replicate_kernel<<<dim3((unsigned)((tot + 255) / 256)), dim3(256), 0, stream>>>(ebuf, erep_bf);
    }
    for (int c = 0; c < MH_ / CH_; c++) {   // per-head adapter, chunked
      launch_gemm(erep_bf + (size_t)c * CH_ * EHD_, a1t[n], (const float*)d_in[pb + 7],
                  nullptr, nullptr, a1_bf, CH_, AH_, EHD_, 1, stream);
      launch_gemm(a1_bf, a2t[n], (const float*)d_in[pb + 9],
                  nullptr, a2buf + (size_t)c * CH_ * HD_, nullptr, CH_, HD_, AH_, 0, stream);
    }
    merge_rope_kernel<<<dim3(T_, NH_), dim3(HD_), 0, stream>>>(
        tbuf, a2buf, ttm, pos, qkv_bf[n], (n < 2) ? 1 : 0);
  }

  // ---- flash attention
  flash_attn_kernel<<<dim3(S_ / 64, NH_, B_), dim3(128), 0, stream>>>(
      qkv_bf[0], qkv_bf[1], qkv_bf[2], o_bf);

  // ---- output projections
  launch_gemm(o_bf, otwt, (const float*)d_in[39], x, outt, nullptr, T_, H_, H_, 0, stream);
  launch_gemm(o_bf, oeint, (const float*)d_in[41], nullptr, e1, nullptr,
              MH_, EHD_, HD_, 0, stream);
  {
    size_t tot = (size_t)T_ * EH_;
    mean_rep_kernel<<<dim3((unsigned)((tot + 255) / 256)), dim3(256), 0, stream>>>(e1, em_bf);
  }
  launch_gemm(em_bf, oewt, (const float*)d_in[43], xe, nullptr, e2_bf, T_, EH_, EH_, 0, stream);
  launch_gemm(e2_bf, oeot, (const float*)d_in[45], nullptr, oute, nullptr, T_, H_, EH_, 0, stream);

  {
    size_t tot = (size_t)T_ * H_;
    final_merge_kernel<<<dim3((unsigned)((tot + 255) / 256)), dim3(256), 0, stream>>>(
        outt, oute, ttm, (float*)d_out);
  }
}
